// RSSM_74594991997110
// MI455X (gfx1250) — matrix operand order repaired and checked by reading, not compiled, not run on hardware
//
#include <hip/hip_runtime.h>
#include <cstdint>
#include <cstddef>

// ---------------------------------------------------------------------------
// RSSM rollout for MI455X (gfx1250, wave32, WMMA).
// B=64 batch rows split across 16 workgroups (4 valid rows each, padded to
// WMMA M=16). Each workgroup runs the full T=256 recurrence persistently,
// state in LDS, weights pre-converted to transposed bf16 in workspace.
// ---------------------------------------------------------------------------

#define T_TOT 256

typedef __bf16 bf16_t;
typedef bf16_t bf16x16 __attribute__((ext_vector_type(16)));
typedef float  f32x8   __attribute__((ext_vector_type(8)));

union ABFrag { bf16x16 v; uint4 q[2]; };

__device__ __forceinline__ unsigned short f2bf(float f) {
  union { float f; unsigned int u; } x; x.f = f;
  unsigned int r = 0x7FFFu + ((x.u >> 16) & 1u);
  return (unsigned short)((x.u + r) >> 16);
}
__device__ __forceinline__ float softplusf_(float x) {
  return (x > 20.f) ? x : log1pf(__expf(x));
}

// ---- WMMA fragment loaders -------------------------------------------------
// A: 16x32 bf16 tile from LDS, row-major with row stride lda (elements).
// lane&15 = M row; lanes<16 hold K {0..7,16..23}, lanes>=16 hold K {8..15,24..31}.
__device__ __forceinline__ bf16x16 load_a_frag(const unsigned short* A, int lda,
                                               int kt, int lane) {
  int m  = lane & 15;
  int kb = kt * 32 + ((lane & 16) ? 8 : 0);
  ABFrag u;
  u.q[0] = *(const uint4*)(A + m * lda + kb);
  u.q[1] = *(const uint4*)(A + m * lda + kb + 16);
  return u.v;
}
// B: 32x16 bf16 tile from global W^T stored [N][Ktot] bf16 row-major.
// lane&15 = N col; lanes<16 hold K {0..7,16..23}, lanes>=16 hold K {8..15,24..31}.
__device__ __forceinline__ bf16x16 load_b_frag(const unsigned short* Wt, int Ktot,
                                               int nt, int kt, int lane) {
  int n  = nt * 16 + (lane & 15);
  int kb = kt * 32 + ((lane & 16) ? 8 : 0);
  const uint4* p = (const uint4*)(Wt + (size_t)n * Ktot + kb);
  ABFrag u;
  u.q[0] = p[0];
  u.q[1] = p[2];
  return u.v;
}
// One 16x16 output tile: acc over nkt K-tiles, optional bias, store f32 to LDS.
// Unroll capped at 4 to keep B-fragment look-ahead within the VGPR budget
// (full unroll caused scratch spills in the GRU K=32 loop).
__device__ __forceinline__ void mm_tile(const unsigned short* A, int lda,
                                        const unsigned short* Wt, int Ktot,
                                        int nt, int nkt,
                                        float* Y, int ldn, const float* bias,
                                        int lane) {
  f32x8 acc = {};
#pragma unroll 4
  for (int kt = 0; kt < nkt; ++kt) {
    bf16x16 a = load_a_frag(A, lda, kt, lane);
    bf16x16 b = load_b_frag(Wt, Ktot, nt, kt, lane);
    acc = __builtin_amdgcn_wmma_f32_16x16x32_bf16(false, a, false, b,
                                                  (short)0, acc, false, false);
  }
  int n  = nt * 16 + (lane & 15);
  int mb = (lane & 16) ? 8 : 0;
  float bv = bias ? bias[n] : 0.f;
#pragma unroll
  for (int k = 0; k < 8; ++k) Y[(mb + k) * ldn + n] = acc[k] + bv;
}

// ---- LayerNorm row stats: warp w reduces row w (wave32 shuffles) -----------
__device__ __forceinline__ void row_stats(const float* Y, int width, float eps,
                                          float* mv, int tid) {
  int w = tid >> 5, lane = tid & 31;
  if (w < 16) {
    float s = 0.f, s2 = 0.f;
    for (int j = lane; j < width; j += 32) {
      float v = Y[w * width + j];
      s += v; s2 += v * v;
    }
#pragma unroll
    for (int m = 16; m >= 1; m >>= 1) {
      s  += __shfl_xor(s,  m, 32);
      s2 += __shfl_xor(s2, m, 32);
    }
    if (lane == 0) {
      float mean = s / width;
      float var  = fmaxf(s2 / width - mean * mean, 0.f);
      mv[w * 2]     = mean;
      mv[w * 2 + 1] = rsqrtf(var + eps);
    }
  }
}

// ---- hidden MLP (Linear -> LN(1e-3) -> SiLU) + stat Linear(512->64, bias) --
__device__ __forceinline__ void hidden_and_stats(
    const unsigned short* A, int lda, int Ktot, int nkt,
    const unsigned short* W1t, const float* lns, const float* lnb,
    const unsigned short* Wst, const float* bst,
    float* s_y, float* s_mv, unsigned short* s_h, float* s_stats, int tid) {
  int wave = tid >> 5, lane = tid & 31;
  mm_tile(A, lda, W1t, Ktot, wave, nkt, s_y, 512, nullptr, lane);  // 32 n-tiles
  __syncthreads();
  row_stats(s_y, 512, 1e-3f, s_mv, tid);
  __syncthreads();
  for (int i = tid; i < 4 * 512; i += 1024) {           // valid rows only
    int r = i >> 9, c = i & 511;
    float v = (s_y[r * 512 + c] - s_mv[r * 2]) * s_mv[r * 2 + 1] * lns[c] + lnb[c];
    v = v / (1.f + __expf(-v));                          // SiLU
    s_h[r * 512 + c] = f2bf(v);
  }
  __syncthreads();
  if (wave < 4)                                          // 64 cols -> 4 tiles
    mm_tile(s_h, 512, Wst, 512, wave, 16, s_stats, 64, bst, lane);
  __syncthreads();
}

// ---- LDS pool layout -------------------------------------------------------
constexpr int OFF_DETER = 0;                     // f32 [16][512]   32768 B
constexpr int OFF_STOCH = OFF_DETER + 32768;     // u16 [16][32]     1024 B
constexpr int OFF_MV    = OFF_STOCH + 1024;      // f32 [16][2]       128 B
constexpr int OFF_STATS = OFF_MV + 128;          // f32 [16][64]     4096 B
constexpr int OFF_Y     = OFF_STATS + 4096;      // f32 [16][512]   32768 B
constexpr int OFF_Y2    = OFF_Y + 32768;         // f32 [16][1536]  98304 B
constexpr int OFF_X     = OFF_Y2 + 98304;        // u16 [16][1536]  49152 B (xin/xg/xo)
constexpr int OFF_H     = OFF_X + 49152;         // u16 [16][512]   16384 B
constexpr int SMEM_BYTES = OFF_H + 16384;        // 234624 B (< 320KB/WGP)

// ---- weight -> transposed bf16 conversion pre-pass -------------------------
__global__ void convert_wT(const float* __restrict__ W, unsigned short* __restrict__ Wt,
                           int K, int N, int Kpad) {
  int idx = blockIdx.x * blockDim.x + threadIdx.x;
  int total = N * Kpad;
  if (idx >= total) return;
  int n = idx / Kpad, k = idx - n * Kpad;
  float v = (k < K) ? W[(size_t)k * N + n] : 0.f;
  Wt[idx] = f2bf(v);
}

// ---- main persistent recurrence kernel -------------------------------------
__global__ __launch_bounds__(1024, 1) void rssm_main(
    const float* __restrict__ embed, const float* __restrict__ context,
    const float* __restrict__ action,
    const float* __restrict__ ln_inp_s, const float* __restrict__ ln_inp_b,
    const float* __restrict__ b_gru,
    const float* __restrict__ ln_gru_s, const float* __restrict__ ln_gru_b,
    const float* __restrict__ ln_img_s, const float* __restrict__ ln_img_b,
    const float* __restrict__ ln_obs_s, const float* __restrict__ ln_obs_b,
    const float* __restrict__ b_ims, const float* __restrict__ b_obs_stat,
    const float* __restrict__ W0,
    const unsigned short* __restrict__ wt_inp, const unsigned short* __restrict__ wt_gru,
    const unsigned short* __restrict__ wt_img, const unsigned short* __restrict__ wt_obs,
    const unsigned short* __restrict__ wt_ims, const unsigned short* __restrict__ wt_stat,
    float* __restrict__ out) {
  extern __shared__ unsigned char smem[];
  float*          s_deter = (float*)(smem + OFF_DETER);
  unsigned short* s_stoch = (unsigned short*)(smem + OFF_STOCH);
  float*          s_mv    = (float*)(smem + OFF_MV);
  float*          s_stats = (float*)(smem + OFF_STATS);
  float*          s_y     = (float*)(smem + OFF_Y);
  float*          s_y2    = (float*)(smem + OFF_Y2);
  unsigned short* s_x     = (unsigned short*)(smem + OFF_X);
  unsigned short* s_h     = (unsigned short*)(smem + OFF_H);

  const int tid  = threadIdx.x;
  const int wave = tid >> 5, lane = tid & 31;
  const int b0   = blockIdx.x * 4;       // first valid batch row of this WG

  // ---- initial state: deter0 = tanh(W0); stoch0 = img_out mean(deter0) ----
  for (int i = tid; i < 16 * 512; i += 1024) {
    int j = i & 511;
    s_deter[i] = tanhf(W0[j]);
  }
  __syncthreads();
  for (int i = tid; i < 4 * 512; i += 1024) {          // deter bf16 at stride 1536
    int r = i >> 9, j = i & 511;
    s_x[r * 1536 + j] = f2bf(s_deter[r * 512 + j]);
  }
  __syncthreads();
  hidden_and_stats(s_x, 1536, 512, 16, wt_img, ln_img_s, ln_img_b,
                   wt_ims, b_ims, s_y, s_mv, s_h, s_stats, tid);
  for (int i = tid; i < 4 * 32; i += 1024) {
    int r = i >> 5, c = i & 31;
    s_stoch[r * 32 + c] = f2bf(s_stats[r * 64 + c]);
  }
  __syncthreads();

  for (int t = 0; t < T_TOT; ++t) {
    // -- phase 1: xin = [stoch | context_t | action_t | 0pad] bf16 [4][576] --
    for (int i = tid; i < 4 * 576; i += 1024) {
      int r = i / 576, c = i - r * 576;
      int b = b0 + r;
      unsigned short o;
      if (c < 32)        o = s_stoch[r * 32 + c];
      else if (c < 544)  o = f2bf(context[((size_t)b * T_TOT + t) * 512 + (c - 32)]);
      else if (c < 550)  o = f2bf(action[((size_t)b * T_TOT + t) * 6 + (c - 544)]);
      else               o = 0;
      s_x[r * 576 + c] = o;
      // prefetch next step's context row into cache (global_prefetch_b8)
      if (t + 1 < T_TOT && c >= 32 && c < 544)
        __builtin_prefetch(&context[((size_t)b * T_TOT + (t + 1)) * 512 + (c - 32)], 0, 3);
    }
    __syncthreads();
    mm_tile(s_x, 576, wt_inp, 576, wave, 18, s_y, 512, nullptr, lane);
    __syncthreads();
    row_stats(s_y, 512, 1e-3f, s_mv, tid);
    __syncthreads();
    // -- xg = [SiLU(LN(y1)) | bf16(deter)] [4][1024] --
    for (int i = tid; i < 4 * 1024; i += 1024) {
      int r = i >> 10, c = i & 1023;
      unsigned short o;
      if (c < 512) {
        float v = (s_y[r * 512 + c] - s_mv[r * 2]) * s_mv[r * 2 + 1] * ln_inp_s[c] + ln_inp_b[c];
        v = v / (1.f + __expf(-v));
        o = f2bf(v);
      } else {
        o = f2bf(s_deter[r * 512 + (c - 512)]);
      }
      s_x[r * 1024 + c] = o;
    }
    __syncthreads();
    // -- GRU linear: [16,1024] @ [1024,1536] + b (96 n-tiles, 3 per wave) --
#pragma unroll
    for (int i = 0; i < 3; ++i)
      mm_tile(s_x, 1024, wt_gru, 1024, wave + 32 * i, 32, s_y2, 1536, b_gru, lane);
    __syncthreads();
    row_stats(s_y2, 1536, 1e-5f, s_mv, tid);
    __syncthreads();
    // -- GRU gates; write new deter (f32 + bf16 copy at stride 1536) --
    for (int i = tid; i < 4 * 512; i += 1024) {
      int r = i >> 9, j = i & 511;
      float mean = s_mv[r * 2], rs = s_mv[r * 2 + 1];
      float xr = (s_y2[r * 1536 + j]        - mean) * rs * ln_gru_s[j]        + ln_gru_b[j];
      float xc = (s_y2[r * 1536 + 512 + j]  - mean) * rs * ln_gru_s[512 + j]  + ln_gru_b[512 + j];
      float xu = (s_y2[r * 1536 + 1024 + j] - mean) * rs * ln_gru_s[1024 + j] + ln_gru_b[1024 + j];
      float reset = 1.f / (1.f + __expf(-xr));
      float cand  = tanhf(reset * xc);
      float upd   = 1.f / (1.f + __expf(-(xu - 1.f)));   // UPDATE_BIAS = -1
      float d = upd * cand + (1.f - upd) * s_deter[r * 512 + j];
      s_deter[r * 512 + j] = d;
      s_x[r * 1536 + j] = f2bf(d);
      // prefetch next step's embed row (consumed in the obs phase)
      if (t + 1 < T_TOT) {
        int b = b0 + r;
        __builtin_prefetch(&embed[((size_t)b * T_TOT + (t + 1)) * 1024 + 2 * j], 0, 3);
      }
    }
    __syncthreads();
    // -- img path: prior stats --
    hidden_and_stats(s_x, 1536, 512, 16, wt_img, ln_img_s, ln_img_b,
                     wt_ims, b_ims, s_y, s_mv, s_h, s_stats, tid);
    // write [p_mean, p_std, p_stoch, deter] at offset 608
    for (int i = tid; i < 4 * 608; i += 1024) {
      int r = i / 608, c = i - r * 608;
      size_t base = ((size_t)(b0 + r) * T_TOT + t) * 1216 + 608;
      float v;
      if (c < 32)       v = s_stats[r * 64 + c];
      else if (c < 64)  v = softplusf_(s_stats[r * 64 + c]) + 0.1f;  // cols 32..63
      else if (c < 96)  v = s_stats[r * 64 + (c - 64)];
      else              v = s_deter[r * 512 + (c - 96)];
      out[base + c] = v;
    }
    // -- xo second half = bf16(embed_t); first half (deter bf16) already set --
    for (int i = tid; i < 4 * 1024; i += 1024) {
      int r = i >> 10, c = i & 1023;
      int b = b0 + r;
      s_x[r * 1536 + 512 + c] = f2bf(embed[((size_t)b * T_TOT + t) * 1024 + c]);
    }
    __syncthreads();
    // -- obs path: posterior stats --
    hidden_and_stats(s_x, 1536, 1536, 48, wt_obs, ln_obs_s, ln_obs_b,
                     wt_stat, b_obs_stat, s_y, s_mv, s_h, s_stats, tid);
    // write [o_mean, o_std, o_stoch, deter] at offset 0; update stoch
    for (int i = tid; i < 4 * 608; i += 1024) {
      int r = i / 608, c = i - r * 608;
      size_t base = ((size_t)(b0 + r) * T_TOT + t) * 1216;
      float v;
      if (c < 32) {
        v = s_stats[r * 64 + c];
        s_stoch[r * 32 + c] = f2bf(v);                   // next-step stoch
      } else if (c < 64) {
        v = softplusf_(s_stats[r * 64 + c]) + 0.1f;
      } else if (c < 96) {
        v = s_stats[r * 64 + (c - 64)];
      } else {
        v = s_deter[r * 512 + (c - 96)];
      }
      out[base + c] = v;
    }
    __syncthreads();
  }
}

// ---- workspace layout (bf16 element offsets) -------------------------------
constexpr size_t WT_INP  = 0;                        // [512][576]
constexpr size_t WT_GRU  = WT_INP  + 512 * 576;      // [1536][1024]
constexpr size_t WT_IMG  = WT_GRU  + 1536 * 1024;    // [512][512]
constexpr size_t WT_OBS  = WT_IMG  + 512 * 512;      // [512][1536]
constexpr size_t WT_IMS  = WT_OBS  + 512 * 1536;     // [64][512]
constexpr size_t WT_STAT = WT_IMS  + 64 * 512;       // [64][512]

extern "C" void kernel_launch(void* const* d_in, const int* in_sizes, int n_in,
                              void* d_out, int out_size, void* d_ws, size_t ws_size,
                              hipStream_t stream) {
  const float* embed      = (const float*)d_in[0];
  const float* context    = (const float*)d_in[1];
  const float* action     = (const float*)d_in[2];
  const float* w_inp      = (const float*)d_in[3];
  const float* ln_inp_s   = (const float*)d_in[4];
  const float* ln_inp_b   = (const float*)d_in[5];
  const float* w_gru      = (const float*)d_in[6];
  const float* b_gru      = (const float*)d_in[7];
  const float* ln_gru_s   = (const float*)d_in[8];
  const float* ln_gru_b   = (const float*)d_in[9];
  const float* w_img      = (const float*)d_in[10];
  const float* ln_img_s   = (const float*)d_in[11];
  const float* ln_img_b   = (const float*)d_in[12];
  const float* w_obs      = (const float*)d_in[13];
  const float* ln_obs_s   = (const float*)d_in[14];
  const float* ln_obs_b   = (const float*)d_in[15];
  const float* w_ims      = (const float*)d_in[16];
  const float* b_ims      = (const float*)d_in[17];
  const float* w_obs_stat = (const float*)d_in[18];
  const float* b_obs_stat = (const float*)d_in[19];
  const float* W0         = (const float*)d_in[20];

  unsigned short* ws = (unsigned short*)d_ws;

  // pre-pass: convert all weights to transposed bf16 [N][Kpad]
  auto launch_conv = [&](const float* W, unsigned short* Wt, int K, int N, int Kpad) {
    int total = N * Kpad;
    convert_wT<<<dim3((total + 255) / 256), dim3(256), 0, stream>>>(W, Wt, K, N, Kpad);
  };
  launch_conv(w_inp,      ws + WT_INP,  550,  512,  576);
  launch_conv(w_gru,      ws + WT_GRU,  1024, 1536, 1024);
  launch_conv(w_img,      ws + WT_IMG,  512,  512,  512);
  launch_conv(w_obs,      ws + WT_OBS,  1536, 512,  1536);
  launch_conv(w_ims,      ws + WT_IMS,  512,  64,   512);
  launch_conv(w_obs_stat, ws + WT_STAT, 512,  64,   512);

  (void)hipFuncSetAttribute((const void*)rssm_main,
                            hipFuncAttributeMaxDynamicSharedMemorySize,
                            (int)SMEM_BYTES);

  rssm_main<<<dim3(16), dim3(1024), SMEM_BYTES, stream>>>(
      embed, context, action,
      ln_inp_s, ln_inp_b, b_gru, ln_gru_s, ln_gru_b,
      ln_img_s, ln_img_b, ln_obs_s, ln_obs_b,
      b_ims, b_obs_stat, W0,
      ws + WT_INP, ws + WT_GRU, ws + WT_IMG,
      ws + WT_OBS, ws + WT_IMS, ws + WT_STAT,
      (float*)d_out);
}
